// E78ProjectedMatrixCell_34059090657894
// MI455X (gfx1250) — matrix operand order repaired and checked by reading, not compiled, not run on hardware
//
#include <hip/hip_runtime.h>
#include <hip/hip_bf16.h>

typedef __attribute__((ext_vector_type(16))) _Float16 v16h;
typedef __attribute__((ext_vector_type(8)))  float    v8f;
typedef __attribute__((ext_vector_type(4)))  float    f4;

#define TT   2048
#define BB   8
#define ROWS (TT*BB)      // 16384
#define DIMK 1024
#define NEFF 256
#define NSM  64
#define NOUT 256          // [k(64) | v(64) | q(64) | gate_raw(64)]

// ---------------------------------------------------------------------------
// 1) Build fused projection weight Mh[n][k] (f16), n in [0,256), k in [0,1024)
//    n<64: (Wk^T P) col s=n ; 64..127: v ; 128..191: q ; 192..255: W_gate row
// ---------------------------------------------------------------------------
__global__ __launch_bounds__(256) void build_Mh(const float* __restrict__ P,
                                                const float* __restrict__ Wkvq,
                                                const float* __restrict__ Wgate,
                                                _Float16* __restrict__ Mh) {
    const int n   = blockIdx.x;      // 0..255
    const int tid = threadIdx.x;     // 0..255
#pragma unroll
    for (int m = 0; m < 4; ++m) {
        const int k = tid + m * 256;
        float acc = 0.0f;
        if (n < 192) {
            const int part = n >> 6;          // 0=k,1=v,2=q
            const int s    = n & 63;
            const float* W = Wkvq + (size_t)part * NEFF * DIMK;
            for (int e = 0; e < NEFF; ++e)
                acc += W[(size_t)e * DIMK + k] * P[e * NSM + s];
        } else {
            acc = Wgate[(size_t)(n - 192) * DIMK + k];
        }
        Mh[(size_t)n * DIMK + k] = (_Float16)acc;
    }
}

// ---------------------------------------------------------------------------
// 2) small[r][n] = x[r][:] . Mh[n][:]   via v_wmma_f32_16x16x32_f16
//    Each wave computes a 16x64 output stripe (4 N-tiles). Software-pipelined:
//    chunk k0+32's A (f32) and 4 B tiles are issued BEFORE the 4 WMMAs of
//    chunk k0, so VMEM latency overlaps matrix work.
// ---------------------------------------------------------------------------
__global__ __launch_bounds__(256) void gemm_small(const float* __restrict__ x,
                                                  const _Float16* __restrict__ Mh,
                                                  float* __restrict__ small) {
    const int wave   = threadIdx.x >> 5;
    const int lane   = threadIdx.x & 31;
    const int tile   = blockIdx.x * 8 + wave;  // 4096 stripe-waves
    const int tileNg = tile & 3;               // 4 N-groups of 64 cols
    const int tileM  = tile >> 2;              // 1024 row tiles (16384/16)
    const int l16    = lane & 15;
    const int hsel   = lane >> 4;              // 0 | 1
    const int ab     = hsel * 16;              // A half-select base
    const int bb     = hsel * 16;              // B half-select base

    const float*    arow = x  + (size_t)(tileM * 16 + l16) * DIMK + ab;
    const _Float16* brow = Mh + (size_t)(tileNg * 64 + l16) * DIMK + bb;

    v8f c0 = {}, c1 = {}, c2 = {}, c3 = {};

    // ---- prologue: load chunk 0 ----
    f4 f0 = *(const f4*)(arow + 0);
    f4 f1 = *(const f4*)(arow + 4);
    f4 f2 = *(const f4*)(arow + 8);
    f4 f3 = *(const f4*)(arow + 12);
    v16h b0 = *(const v16h*)(brow + 0 * 16 * DIMK);
    v16h b1 = *(const v16h*)(brow + 1 * 16 * DIMK);
    v16h b2 = *(const v16h*)(brow + 2 * 16 * DIMK);
    v16h b3 = *(const v16h*)(brow + 3 * 16 * DIMK);

    for (int k0 = 0; k0 < DIMK - 32; k0 += 32) {
        const int kn = k0 + 32;
        // ---- issue next chunk's loads first (overlap with WMMAs below) ----
        f4 nf0 = *(const f4*)(arow + kn);
        f4 nf1 = *(const f4*)(arow + kn + 4);
        f4 nf2 = *(const f4*)(arow + kn + 8);
        f4 nf3 = *(const f4*)(arow + kn + 12);
        v16h nb0 = *(const v16h*)(brow + 0 * 16 * DIMK + kn);
        v16h nb1 = *(const v16h*)(brow + 1 * 16 * DIMK + kn);
        v16h nb2 = *(const v16h*)(brow + 2 * 16 * DIMK + kn);
        v16h nb3 = *(const v16h*)(brow + 3 * 16 * DIMK + kn);
        __builtin_prefetch(arow + k0 + 160, 0, 1);   // A chunk five ahead (k0+160)

        // ---- convert current A tile f32 -> f16 ----
        v16h a;
#pragma unroll
        for (int u = 0; u < 4; ++u) {
            a[u]      = (_Float16)f0[u];
            a[4 + u]  = (_Float16)f1[u];
            a[8 + u]  = (_Float16)f2[u];
            a[12 + u] = (_Float16)f3[u];
        }
        // ---- 4 WMMAs on current chunk ----
        c0 = __builtin_amdgcn_wmma_f32_16x16x32_f16(false, a, false, b0,
                                                    (short)0, c0, false, false);
        c1 = __builtin_amdgcn_wmma_f32_16x16x32_f16(false, a, false, b1,
                                                    (short)0, c1, false, false);
        c2 = __builtin_amdgcn_wmma_f32_16x16x32_f16(false, a, false, b2,
                                                    (short)0, c2, false, false);
        c3 = __builtin_amdgcn_wmma_f32_16x16x32_f16(false, a, false, b3,
                                                    (short)0, c3, false, false);
        // ---- rotate buffers ----
        f0 = nf0; f1 = nf1; f2 = nf2; f3 = nf3;
        b0 = nb0; b1 = nb1; b2 = nb2; b3 = nb3;
    }
    // ---- epilogue: last chunk ----
    {
        v16h a;
#pragma unroll
        for (int u = 0; u < 4; ++u) {
            a[u]      = (_Float16)f0[u];
            a[4 + u]  = (_Float16)f1[u];
            a[8 + u]  = (_Float16)f2[u];
            a[12 + u] = (_Float16)f3[u];
        }
        c0 = __builtin_amdgcn_wmma_f32_16x16x32_f16(false, a, false, b0,
                                                    (short)0, c0, false, false);
        c1 = __builtin_amdgcn_wmma_f32_16x16x32_f16(false, a, false, b1,
                                                    (short)0, c1, false, false);
        c2 = __builtin_amdgcn_wmma_f32_16x16x32_f16(false, a, false, b2,
                                                    (short)0, c2, false, false);
        c3 = __builtin_amdgcn_wmma_f32_16x16x32_f16(false, a, false, b3,
                                                    (short)0, c3, false, false);
    }

    // D: VGPR j -> row (tileM*16 + hsel*8 + j), col (tileNg*64 + sub*16 + l16)
    float* orow = small + (size_t)(tileM * 16 + hsel * 8) * NOUT
                        + tileNg * 64 + l16;
#pragma unroll
    for (int j = 0; j < 8; ++j) {
        orow[(size_t)j * NOUT + 0]  = c0[j];
        orow[(size_t)j * NOUT + 16] = c1[j];
        orow[(size_t)j * NOUT + 32] = c2[j];
        orow[(size_t)j * NOUT + 48] = c3[j];
    }
}

// ---------------------------------------------------------------------------
// 3) Sequential scan. Rows of S are independent: one wave32 handles
//    (batch b, 8 rows); 4 lanes per row, 16 S-elements per lane, in registers.
//    Row reductions via __shfl_xor(1),(2) within the 4-lane group. No LDS.
// ---------------------------------------------------------------------------
__global__ __launch_bounds__(32) void scan_kernel(const float* __restrict__ small,
                                                  const float* __restrict__ S0,
                                                  const float* __restrict__ bgate,
                                                  float* __restrict__ Sq_small,
                                                  float* __restrict__ S_final) {
    const int b    = blockIdx.x >> 3;          // 0..7
    const int rg   = blockIdx.x & 7;           // row group
    const int lane = threadIdx.x;
    const int i    = rg * 8 + (lane >> 2);     // state row 0..63
    const int j0   = (lane & 3) * 16;          // column segment

    float S[16];
    {
        const float* s0 = S0 + ((size_t)b * 64 + i) * 64 + j0;
#pragma unroll
        for (int c = 0; c < 16; ++c) S[c] = s0[c];
    }
    const float bias = bgate[i];

    for (int t = 0; t < TT; ++t) {
        const float* row = small + (size_t)(t * BB + b) * NOUT;
        float kv[16], qv[16];
        {
            f4 a0 = *(const f4*)(row + j0);
            f4 a1 = *(const f4*)(row + j0 + 4);
            f4 a2 = *(const f4*)(row + j0 + 8);
            f4 a3 = *(const f4*)(row + j0 + 12);
            f4 b0 = *(const f4*)(row + 128 + j0);
            f4 b1 = *(const f4*)(row + 128 + j0 + 4);
            f4 b2 = *(const f4*)(row + 128 + j0 + 8);
            f4 b3 = *(const f4*)(row + 128 + j0 + 12);
#pragma unroll
            for (int u = 0; u < 4; ++u) {
                kv[u] = a0[u]; kv[4+u] = a1[u]; kv[8+u] = a2[u]; kv[12+u] = a3[u];
                qv[u] = b0[u]; qv[4+u] = b1[u]; qv[8+u] = b2[u]; qv[12+u] = b3[u];
            }
        }
        const float vi   = row[64 + i];
        const float graw = row[192 + i];
        const float dec  = 1.0f / (1.0f + __expf(-(graw + bias)));

        // ||k||^2 over all 64 j (4-lane group spans the full row)
        float ss = 0.0f;
#pragma unroll
        for (int c = 0; c < 16; ++c) ss += kv[c] * kv[c];
        ss += __shfl_xor(ss, 1);
        ss += __shfl_xor(ss, 2);
        const float inv = 1.0f / (sqrtf(ss) + 1e-6f);

        // retrieved_i = inv * sum_j S_ij k_j
        float r = 0.0f;
#pragma unroll
        for (int c = 0; c < 16; ++c) r += S[c] * kv[c];
        r += __shfl_xor(r, 1);
        r += __shfl_xor(r, 2);
        const float delta = vi - r * inv;

        // S update + Sq partial
        float sq = 0.0f;
#pragma unroll
        for (int c = 0; c < 16; ++c) {
            const float kn = kv[c] * inv;
            S[c] = dec * S[c] + delta * kn;
            sq  += S[c] * qv[c];
        }
        sq += __shfl_xor(sq, 1);
        sq += __shfl_xor(sq, 2);
        if ((lane & 3) == 0)
            Sq_small[(size_t)(t * BB + b) * NSM + i] = sq;
    }

    float* sf = S_final + ((size_t)b * 64 + i) * 64 + j0;
#pragma unroll
    for (int c = 0; c < 16; ++c) sf[c] = S[c];
}

// ---------------------------------------------------------------------------
// 4) out[r][e] = y*silu(y),  y = Sq_small[r][:] . P[e][:]   (K=64)
// ---------------------------------------------------------------------------
__global__ __launch_bounds__(256) void out_kernel(const float* __restrict__ Sq_small,
                                                  const float* __restrict__ P,
                                                  float* __restrict__ out) {
    __shared__ float srow[NSM];
    const int r = blockIdx.x;          // 0..16383
    const int e = threadIdx.x;         // 0..255
    if (e < NSM) srow[e] = Sq_small[(size_t)r * NSM + e];
    __syncthreads();
    const float* pe = P + (size_t)e * NSM;
    float acc = 0.0f;
#pragma unroll
    for (int s = 0; s < NSM; s += 4) {
        f4 p4 = *(const f4*)(pe + s);
        acc += p4[0] * srow[s] + p4[1] * srow[s + 1]
             + p4[2] * srow[s + 2] + p4[3] * srow[s + 3];
    }
    const float sg = 1.0f / (1.0f + __expf(-acc));
    out[(size_t)r * NEFF + e] = acc * acc * sg;   // y * silu(y)
}

// ---------------------------------------------------------------------------
extern "C" void kernel_launch(void* const* d_in, const int* in_sizes, int n_in,
                              void* d_out, int out_size, void* d_ws, size_t ws_size,
                              hipStream_t stream) {
    const float* x     = (const float*)d_in[0];
    const float* S0    = (const float*)d_in[1];
    const float* P     = (const float*)d_in[2];
    const float* Wkvq  = (const float*)d_in[3];
    const float* Wgate = (const float*)d_in[4];
    const float* bgate = (const float*)d_in[5];
    float* out = (float*)d_out;

    char* ws = (char*)d_ws;
    _Float16* Mh    = (_Float16*)ws;                                   // 512 KB
    float* small    = (float*)(ws + (512 << 10));                      // 16.8 MB
    float* Sq_small = (float*)(ws + (512 << 10) + (size_t)ROWS * NOUT * 4); // 4.2 MB
    float* S_final  = out + (size_t)ROWS * NEFF;                       // output tail

    build_Mh  <<<256,   256, 0, stream>>>(P, Wkvq, Wgate, Mh);
    gemm_small<<<512,   256, 0, stream>>>(x, Mh, small);
    scan_kernel<<<64,    32, 0, stream>>>(small, S0, bgate, Sq_small, S_final);
    out_kernel<<<ROWS,  256, 0, stream>>>(Sq_small, P, out);
}
